// LengthRegulator_7962869367521
// MI455X (gfx1250) — matrix operand order repaired and checked by reading, not compiled, not run on hardware
//
#include <hip/hip_runtime.h>
#include <hip/hip_bf16.h>

typedef __attribute__((ext_vector_type(16))) _Float16 v16h;
typedef __attribute__((ext_vector_type(8)))  _Float16 v8h;
typedef __attribute__((ext_vector_type(8)))  float    v8f;

constexpr int Bc  = 32;
constexpr int Tc  = 512;   // phoneme axis (K of GEMM)
constexpr int Dc  = 512;   // embedding axis (N of GEMM)
constexpr int TSc = 2304;  // frame axis T* = TL-1 (M of GEMM)
constexpr int WPAD = 520;  // padded LDS row stride (halves); 520*2=1040B, 16B-aligned rows

// ---------------- kernel 1: cumsum / softplus / centers -------------------
__global__ void lr_setup(const int* __restrict__ ds, const float* __restrict__ sigma,
                         float* __restrict__ center, float* __restrict__ sig,
                         int* __restrict__ total) {
  __shared__ int dsRow[Tc];
  __shared__ int lenRow[Tc];
  const int b = blockIdx.x;
  const int tid = threadIdx.x;
  for (int i = tid; i < Tc; i += blockDim.x) dsRow[i] = ds[b * Tc + i];
  __syncthreads();
  if (tid == 0) {
    int acc = 0;
    for (int i = 0; i < Tc; ++i) { acc += dsRow[i]; lenRow[i] = acc; }
    total[b] = acc;
  }
  __syncthreads();
  for (int i = tid; i < Tc; i += blockDim.x) {
    const int l = lenRow[i], d = dsRow[i];
    const float shifted = (float)(l - d);
    center[b * Tc + i] = shifted + 0.5f * (float)d;
    const float x = sigma[b * Tc + i];
    // softplus(x) = max(x,0) + log1p(exp(-|x|))  (overflow-safe)
    sig[b * Tc + i] = fmaxf(x, 0.0f) + log1pf(expf(-fabsf(x)));
  }
}

// ---------------- kernel 2: xs (B,T,D) f32 -> xsT (B,D,T) f16 -------------
__global__ void lr_transpose(const float* __restrict__ xs, _Float16* __restrict__ xsT) {
  __shared__ float tile[16][17];
  const int b  = blockIdx.z;
  const int s0 = blockIdx.y * 16;
  const int d0 = blockIdx.x * 16;
  const int tx = threadIdx.x, ty = threadIdx.y;
  tile[ty][tx] = xs[((size_t)b * Tc + (s0 + ty)) * Dc + (d0 + tx)];
  __syncthreads();
  xsT[((size_t)b * Dc + (d0 + ty)) * Tc + (s0 + tx)] = (_Float16)tile[tx][ty];
}

// ---------------- kernel 3: build w-tile + WMMA GEMM ----------------------
__global__ __launch_bounds__(256) void lr_main(const _Float16* __restrict__ xsT,
                                               const float* __restrict__ center,
                                               const float* __restrict__ sig,
                                               const int* __restrict__ total,
                                               float* __restrict__ out) {
  __shared__ float cenL[Tc];
  __shared__ float sigL[Tc];
  __shared__ __align__(32) _Float16 wL[16 * WPAD];

  const int tilesPerB = TSc / 16;                 // 144
  const int b  = blockIdx.x / tilesPerB;
  const int t0 = (blockIdx.x % tilesPerB) * 16;
  const int tid  = threadIdx.x;
  const int lane = tid & 31;
  const int wave = tid >> 5;                      // 8 waves

  for (int i = tid; i < Tc; i += 256) {
    cenL[i] = center[b * Tc + i];
    sigL[i] = sig[b * Tc + i];
  }
  const int tot = total[b];
  __syncthreads();

  // ---- phase 1: each wave builds 2 rows of normalized w (f16) in LDS ----
  for (int rr = 0; rr < 2; ++rr) {
    const int r = wave * 2 + rr;
    const float tr = (float)(t0 + r + 1);         // 1-based frame index
    float vals[16];
    float s = 0.0f;
#pragma unroll
    for (int m = 0; m < 16; ++m) {
      const int i = lane + 32 * m;                // conflict-free LDS reads
      const float sg = sigL[i];
      const float dd = tr - cenL[i];
      const float e  = __expf(-0.5f * dd * dd * sg * sg) * sg;
      vals[m] = e;
      s += e;
    }
#pragma unroll
    for (int off = 16; off > 0; off >>= 1) s += __shfl_xor(s, off, 32);
    const float rowmass = ((t0 + r) < tot) ? 1.0f : 0.0f;   // exact row mass
    const float scale = rowmass / fmaxf(s, 1e-12f);
#pragma unroll
    for (int m = 0; m < 16; ++m)
      wL[r * WPAD + lane + 32 * m] = (_Float16)(vals[m] * scale);
  }
  __syncthreads();

  // ---- phase 2: 16x512 @ 512x512 via v_wmma_f32_16x16x32_f16 ------------
  const int row = lane & 15;
  const int hi  = lane >> 4;
  v8f acc[4] = {};                                // 4 N-tiles per wave
  const _Float16* xb = xsT + (size_t)b * Dc * Tc;

  // B fragment loader: lane holds 16 contiguous K at column N = d0 + row
  auto loadB = [&](int k0, int q) -> v16h {
    const int d0 = (wave * 4 + q) * 16;
    return *(const v16h*)(xb + (size_t)(d0 + row) * Tc + k0 + 16 * hi);
  };

  // software pipeline: B fragments double-buffered in registers so loads
  // for k0+32 are in flight while the WMMAs for k0 execute
  v16h bcur[4];
#pragma unroll
  for (int q = 0; q < 4; ++q) bcur[q] = loadB(0, q);

#pragma unroll
  for (int k0 = 0; k0 < Tc; k0 += 32) {
    // A fragment (contiguous K, matching B): lanes 0-15: K=k0..k0+15
    //                                        lanes 16-31: K=k0+16..k0+31
    const v8h a_lo = *(const v8h*)(&wL[row * WPAD + k0 + 16 * hi]);
    const v8h a_hi = *(const v8h*)(&wL[row * WPAD + k0 + 16 * hi + 8]);
    const v16h A = __builtin_shufflevector(a_lo, a_hi,
        0, 1, 2, 3, 4, 5, 6, 7, 8, 9, 10, 11, 12, 13, 14, 15);

    v16h bnext[4];
    if (k0 + 32 < Tc) {
#pragma unroll
      for (int q = 0; q < 4; ++q) bnext[q] = loadB(k0 + 32, q);
    }

#pragma unroll
    for (int q = 0; q < 4; ++q)
      acc[q] = __builtin_amdgcn_wmma_f32_16x16x32_f16(
          false, A, false, bcur[q], (short)0, acc[q], false, false);

    if (k0 + 32 < Tc) {
#pragma unroll
      for (int q = 0; q < 4; ++q) bcur[q] = bnext[q];
    }
  }

  // ---- store: f32 C/D layout: lane -> N = lane&15, VGPR j -> M = j + 8*hi
#pragma unroll
  for (int q = 0; q < 4; ++q) {
    const int d0 = (wave * 4 + q) * 16;
    float* op = out + ((size_t)b * TSc + t0 + 8 * hi) * Dc + d0 + row;
#pragma unroll
    for (int j = 0; j < 8; ++j) op[(size_t)j * Dc] = acc[q][j];
  }
}

// ---------------------------------------------------------------------------
extern "C" void kernel_launch(void* const* d_in, const int* in_sizes, int n_in,
                              void* d_out, int out_size, void* d_ws, size_t ws_size,
                              hipStream_t stream) {
  const float* xs    = (const float*)d_in[0];   // (B,T,D) f32
  const int*   ds    = (const int*)d_in[1];     // (B,T) i32
  const float* sigma = (const float*)d_in[2];   // (B,T) f32
  (void)in_sizes; (void)n_in; (void)out_size; (void)ws_size;

  // workspace layout
  char* ws = (char*)d_ws;
  _Float16* xsT = (_Float16*)ws;                               // 16 MiB
  size_t off = (size_t)Bc * Dc * Tc * sizeof(_Float16);
  float* center = (float*)(ws + off);  off += (size_t)Bc * Tc * sizeof(float);
  float* sig    = (float*)(ws + off);  off += (size_t)Bc * Tc * sizeof(float);
  int*   total  = (int*)(ws + off);

  lr_setup<<<Bc, 256, 0, stream>>>(ds, sigma, center, sig, total);
  lr_transpose<<<dim3(Dc / 16, Tc / 16, Bc), dim3(16, 16), 0, stream>>>(xs, xsT);
  lr_main<<<Bc * (TSc / 16), 256, 0, stream>>>(xsT, center, sig, total, (float*)d_out);
}
